// SWAN_11123965296690
// MI455X (gfx1250) — matrix operand order repaired and checked by reading, not compiled, not run on hardware
//
#include <hip/hip_runtime.h>
#include <hip/hip_bf16.h>
#include <math.h>

typedef __attribute__((ext_vector_type(16))) __bf16 v16bf;
typedef __attribute__((ext_vector_type(8)))  __bf16 v8bf;
typedef __attribute__((ext_vector_type(8)))  float  v8f;
typedef __attribute__((ext_vector_type(4)))  int    v4i;

namespace {
constexpr int V_ = 8000, E_ = 256, H_ = 512, OUT_ = 256, L_ = 2, U_ = 1024;
constexpr int B_ = 8, T1_ = 32, T2_ = 32;
constexpr int J_ = T2_ + 1;        // 33
constexpr int KSTEPS_ = 9;         // MAXSEG+1
constexpr int N_ = B_ * T1_ * J_;  // 8448 (132 * 64)
constexpr int FH_ = 4 * H_;        // 2048
constexpr int START_ID_ = V_ + 1;
constexpr int EOS_ = 2;
constexpr float NEGV = -1000000000.0f;
}

#define CDIV(a,b) (((a)+(b)-1)/(b))

// ---------------------------------------------------------------------------
// Async global->LDS staging (CDNA5 GLOBAL_LOAD_ASYNC_TO_LDS_B128, ASYNCcnt).
// Builtin signature (from compiler diagnostic): (v4i as1*, v4i as3*, imm, imm).
// Guarded: falls back to batched VGPR round-trip copies if not declared.
// ---------------------------------------------------------------------------
#if defined(__AMDGCN__) && \
    __has_builtin(__builtin_amdgcn_global_load_async_to_lds_b128) && \
    __has_builtin(__builtin_amdgcn_s_wait_asynccnt)
#define HAVE_ASYNC_LDS 1
typedef __attribute__((address_space(1))) v4i as1_v4i;
typedef __attribute__((address_space(3))) v4i as3_v4i;
#else
#define HAVE_ASYNC_LDS 0
#endif

__device__ __forceinline__ void stage16(const __bf16* g, __bf16* l) {
#if HAVE_ASYNC_LDS
  __builtin_amdgcn_global_load_async_to_lds_b128((as1_v4i*)g, (as3_v4i*)l, 0, 0);
#else
  *(v8bf*)l = *(const v8bf*)g;
#endif
}

__device__ __forceinline__ void stage_fence() {
#if HAVE_ASYNC_LDS
  __builtin_amdgcn_s_wait_asynccnt(0);
#endif
  __syncthreads();
}

// ---------------------------------------------------------------------------
// WMMA helper: D = A(16x32 bf16) * B(32x16 bf16) + C (f32)
// ---------------------------------------------------------------------------
__device__ __forceinline__ v8f wmma_bf16(v16bf a, v16bf b, v8f c) {
  return __builtin_amdgcn_wmma_f32_16x16x32_bf16(false, a, false, b, (short)0, c,
                                                 false, false);
}

// A fragment (16x32 bf16): lane half h, e<8 -> K = 8h+e ; e>=8 -> K = 16+8h+(e-8)
__device__ __forceinline__ v16bf make_a_frag(const __bf16* __restrict__ rowK, int half) {
  v16bf a;
#pragma unroll
  for (int e = 0; e < 8; ++e) {
    a[e]     = rowK[half * 8 + e];
    a[8 + e] = rowK[16 + half * 8 + e];
  }
  return a;
}

// B fragment (32x16 bf16): lane col N = ml, e<8 -> K = 8h+e ; e>=8 -> K = 16+8h+(e-8).
__device__ __forceinline__ v16bf make_b_frag(const __bf16* __restrict__ colK, int half) {
  v16bf b;
#pragma unroll
  for (int e = 0; e < 8; ++e) {
    b[e]     = colK[half * 8 + e];
    b[8 + e] = colK[16 + half * 8 + e];
  }
  return b;
}

// ---------------------------------------------------------------------------
// Fused recurrent-gate GEMM + LSTM cell.
//   G[n, 4H] = A1[n,K1]@B1^T (+ A2[n,K2]@B2^T) + addend ; then cell update.
// Block: 128 threads = 4 waves. Tile: 64 rows x 16 h-cols, all 4 gate groups.
// K1, K2 multiples of 32 so chunks never straddle regions.
// mode 0 (layer0): addend = gembS[(b*J+j)*4H+g] + gctx[(b*T1+t1)*4H+g]
// mode 1 (layer1): addend = bias1[g]
// ---------------------------------------------------------------------------
__global__ __launch_bounds__(128) void k_gates(
    const __bf16* __restrict__ A1, int K1,
    const __bf16* __restrict__ A2, int K2,
    const __bf16* __restrict__ B1,
    const __bf16* __restrict__ B2,
    const float*  __restrict__ addA,
    const float*  __restrict__ addB,
    float*        __restrict__ Cst,   // c state (in/out, elementwise exact)
    __bf16*       __restrict__ Hout,  // new h state (bf16 for next GEMM)
    int mode)
{
  __shared__ alignas(16) __bf16 As[64][40];      // pad 32->40: conflict-free frags
  __shared__ alignas(16) __bf16 Bs[4][16][40];

  const int tid  = threadIdx.x;
  const int wave = tid >> 5, lane = tid & 31, half = lane >> 4, ml = lane & 15;
  const int rowbase = blockIdx.x * 64;
  const int c0col   = blockIdx.y * 16;
  const int Ktot = K1 + K2;

  v8f acc[4];
#pragma unroll
  for (int q = 0; q < 4; ++q)
#pragma unroll
    for (int e = 0; e < 8; ++e) acc[q][e] = 0.f;

  for (int kk = 0; kk < Ktot; kk += 32) {
    // Region select is uniform per chunk: pure 32-bit indexing below.
    const __bf16* Asrc; const __bf16* Bsrc; int Kst;
    if (kk < K1) { Asrc = A1 + kk;        Bsrc = B1 + kk;        Kst = K1; }
    else         { Asrc = A2 + (kk - K1); Bsrc = B2 + (kk - K1); Kst = K2; }

#pragma unroll
    for (int t = 0; t < 2; ++t) {           // A tile: 64x32 = 256 x 16B
      int idx = tid + t * 128;
      int r = idx >> 2, c8 = (idx & 3) << 3;
      stage16(Asrc + (rowbase + r) * Kst + c8, &As[r][c8]);
    }
#pragma unroll
    for (int t = 0; t < 2; ++t) {           // B tiles: 4x16x32 = 256 x 16B
      int idx = tid + t * 128;
      int q = idx >> 6, rest = idx & 63;
      int col = rest >> 2, c8 = (rest & 3) << 3;
      stage16(Bsrc + (q * H_ + c0col + col) * Kst + c8, &Bs[q][col][c8]);
    }
    stage_fence();

    v16bf a = make_a_frag(&As[wave * 16 + ml][0], half);
    v16bf bf[4];
#pragma unroll
    for (int q = 0; q < 4; ++q) bf[q] = make_b_frag(&Bs[q][ml][0], half);
#pragma unroll
    for (int q = 0; q < 4; ++q) acc[q] = wmma_bf16(a, bf[q], acc[q]);
    __syncthreads();
  }

  // Epilogue: lane-local LSTM cell. C layout: row M = r + 8*half, col = ml.
  const int hcol = c0col + ml;
#pragma unroll
  for (int r = 0; r < 8; ++r) {
    int n = rowbase + wave * 16 + r + 8 * half;
    float gv[4];
#pragma unroll
    for (int q = 0; q < 4; ++q) {
      int gcol = q * H_ + hcol;
      float add;
      if (mode == 0) {
        int b = n / (T1_ * J_); int rem = n % (T1_ * J_);
        int t1 = rem / J_;      int j   = rem % J_;
        add = addA[(b * J_ + j) * FH_ + gcol] + addB[(b * T1_ + t1) * FH_ + gcol];
      } else {
        add = addA[gcol];
      }
      gv[q] = acc[q][r] + add;
    }
    float cold = Cst[n * H_ + hcol];
    float ig = 1.f / (1.f + __expf(-gv[0]));
    float fg = 1.f / (1.f + __expf(-gv[1]));
    float gg = tanhf(gv[2]);
    float og = 1.f / (1.f + __expf(-gv[3]));
    float cnew = fg * cold + ig * gg;
    float hnew = og * tanhf(cnew);
    Cst[n * H_ + hcol]  = cnew;
    Hout[n * H_ + hcol] = (__bf16)hnew;
  }
}

// ---------------------------------------------------------------------------
// fc GEMM: O[n,256] = h1n[n,512] @ Wfc^T + bfc, written as bf16 for logits GEMM.
// Block tile 64 rows x 64 cols (4 col-groups of 16), 4 waves, K = 512.
// ---------------------------------------------------------------------------
__global__ __launch_bounds__(128) void k_fc(
    const __bf16* __restrict__ A,     // N x 512
    const __bf16* __restrict__ Bw,    // 256 x 512 (row-major)
    const float*  __restrict__ bias,  // 256
    __bf16*       __restrict__ Obf)   // N x 256
{
  __shared__ alignas(16) __bf16 As[64][40];
  __shared__ alignas(16) __bf16 Bs[4][16][40];

  const int tid  = threadIdx.x;
  const int wave = tid >> 5, lane = tid & 31, half = lane >> 4, ml = lane & 15;
  const int rowbase = blockIdx.x * 64;
  const int cbase   = blockIdx.y * 64;

  v8f acc[4];
#pragma unroll
  for (int q = 0; q < 4; ++q)
#pragma unroll
    for (int e = 0; e < 8; ++e) acc[q][e] = 0.f;

  for (int kk = 0; kk < H_; kk += 32) {
#pragma unroll
    for (int t = 0; t < 2; ++t) {
      int idx = tid + t * 128;
      int r = idx >> 2, c8 = (idx & 3) << 3;
      stage16(A + (rowbase + r) * H_ + kk + c8, &As[r][c8]);
    }
#pragma unroll
    for (int t = 0; t < 2; ++t) {
      int idx = tid + t * 128;
      int q = idx >> 6, rest = idx & 63;
      int col = rest >> 2, c8 = (rest & 3) << 3;
      stage16(Bw + (cbase + q * 16 + col) * H_ + kk + c8, &Bs[q][col][c8]);
    }
    stage_fence();

    v16bf a = make_a_frag(&As[wave * 16 + ml][0], half);
    v16bf bf[4];
#pragma unroll
    for (int q = 0; q < 4; ++q) bf[q] = make_b_frag(&Bs[q][ml][0], half);
#pragma unroll
    for (int q = 0; q < 4; ++q) acc[q] = wmma_bf16(a, bf[q], acc[q]);
    __syncthreads();
  }

#pragma unroll
  for (int q = 0; q < 4; ++q) {
    int colg = cbase + q * 16 + ml;
    float bv = bias[colg];
#pragma unroll
    for (int r = 0; r < 8; ++r) {
      int n = rowbase + wave * 16 + r + 8 * half;
      Obf[n * OUT_ + colg] = (__bf16)(acc[q][r] + bv);
    }
  }
}

// ---------------------------------------------------------------------------
// Fused logits + log-softmax (online, lane-local). Block = 4 waves x 16 rows.
// Wout tile (16 vocab rows x 256 K) staged in LDS once per chunk, shared by
// all 4 waves (4x L2 traffic cut). Per-lane running (max,sum): lane ml owns
// logit column v0+ml for its 8 C-rows -> zero shuffles in the hot loop.
// EOS/target logits captured by the owning lane; merged once at the end.
// ---------------------------------------------------------------------------
__global__ __launch_bounds__(128) void k_logits(
    const __bf16* __restrict__ Obf,     // N x 256
    const __bf16* __restrict__ WoutBf,  // 8000 x 256
    const float*  __restrict__ bout,    // 8000
    const int*    __restrict__ prev,    // B x T2
    float*        __restrict__ cum,     // N
    float*        __restrict__ outp,    // (B,T1,J,K)
    int kstep)
{
  __shared__ alignas(16) __bf16 Ws[16][264];   // pad 256->264 (4-bank row skew)
  __shared__ float Bo[16];

  const int tid  = threadIdx.x;
  const int wave = tid >> 5, lane = tid & 31, half = lane >> 4, ml = lane & 15;
  const int rb = blockIdx.x * 64 + wave * 16;

  // A fragments for this wave's 16x256 tile of O (held in registers).
  v16bf a[8];
  {
    const __bf16* rp = Obf + (rb + ml) * OUT_;
#pragma unroll
    for (int kc = 0; kc < 8; ++kc) {
#pragma unroll
      for (int e = 0; e < 8; ++e) {
        a[kc][e]     = rp[kc * 32 + half * 8 + e];
        a[kc][8 + e] = rp[kc * 32 + 16 + half * 8 + e];
      }
    }
  }

  float m[8], s[8], ecap[8], tcap[8];
  int tgt[8], rowI[8], oIdx[8], vld[8];
#pragma unroll
  for (int r = 0; r < 8; ++r) {
    int n = rb + r + 8 * half;
    rowI[r] = n;
    int b = n / (T1_ * J_); int rem = n % (T1_ * J_);
    int t1 = rem / J_;      int j   = rem % J_;
    int tpos = j + kstep;
    int cp = (tpos > T2_ - 1) ? (T2_ - 1) : tpos;
    tgt[r]  = prev[b * T2_ + cp];
    vld[r]  = (tpos <= T2_) ? 1 : 0;
    oIdx[r] = ((b * T1_ + t1) * J_ + j) * KSTEPS_ + kstep;
    m[r] = -3.0e38f; s[r] = 0.f; ecap[r] = 0.f; tcap[r] = 0.f;
  }

  for (int v0 = 0; v0 < V_; v0 += 16) {
    __syncthreads();   // protect LDS WAR from previous chunk's readers
#pragma unroll
    for (int t = 0; t < 4; ++t) {           // 16x256 = 512 x 16B
      int idx = tid + t * 128;
      int row = idx >> 5, c8 = (idx & 31) << 3;
      const __bf16* src = WoutBf + (v0 + row) * OUT_ + c8;
      stage16(src, &Ws[row][c8]);
      __builtin_prefetch(src + 16 * OUT_, 0, 3);   // next chunk -> global_prefetch_b8
    }
    if (tid < 16) Bo[tid] = bout[v0 + tid];
    stage_fence();

    v8f c;
#pragma unroll
    for (int e = 0; e < 8; ++e) c[e] = 0.f;
    const __bf16* wr = &Ws[ml][0];
#pragma unroll
    for (int kc = 0; kc < 8; ++kc) {
      v16bf b = make_b_frag(wr + kc * 32, half);
      c = wmma_bf16(a[kc], b, c);
    }

    float bo = Bo[ml];
    int vcol = v0 + ml;
#pragma unroll
    for (int r = 0; r < 8; ++r) {
      float val = c[r] + bo;   // logit(row rb+r+8*half, col vcol)
      float nm = fmaxf(m[r], val);
      s[r] = s[r] * __expf(m[r] - nm) + __expf(val - nm);
      m[r] = nm;
      if (vcol == EOS_)   ecap[r] = val;
      if (vcol == tgt[r]) tcap[r] = val;
    }
  }

  // Merge per-lane (max,sum) across the 16 lanes of each half; emit results.
#pragma unroll
  for (int r = 0; r < 8; ++r) {
    float mm = m[r], ss = s[r];
#pragma unroll
    for (int off = 1; off < 16; off <<= 1) {
      float om = __shfl_xor(mm, off, 32);
      float os = __shfl_xor(ss, off, 32);
      float nm = fmaxf(mm, om);
      ss = ss * __expf(mm - nm) + os * __expf(om - nm);
      mm = nm;
    }
    float lse  = mm + __logf(ss);
    float eosv = __shfl(ecap[r], half * 16 + EOS_, 32);
    float tgtv = __shfl(tcap[r], half * 16 + (tgt[r] & 15), 32);
    float cu = cum[rowI[r]];
    float ly = vld[r] ? (cu + eosv - lse) : NEGV;
    float nc = cu + tgtv - lse;
    if (ml == 0) { outp[oIdx[r]] = ly; cum[rowI[r]] = nc; }
  }
}

// ---------------------------------------------------------------------------
// Small one-time / per-step helper kernels (negligible FLOPs vs the WMMA GEMMs)
// ---------------------------------------------------------------------------
__global__ void k_cvt(const float* __restrict__ s, __bf16* __restrict__ d, int n) {
  int i = blockIdx.x * blockDim.x + threadIdx.x;
  if (i < n) d[i] = (__bf16)s[i];
}

__global__ void k_bias1(const float* __restrict__ a, const float* __restrict__ b,
                        float* __restrict__ d) {
  int i = blockIdx.x * blockDim.x + threadIdx.x;
  if (i < FH_) d[i] = a[i] + b[i];
}

// h0i/c0i: (L,B,H) projections of encoder hidden/cell states.
__global__ void k_proj_hc(const float* __restrict__ encH, const float* __restrict__ encC,
                          const float* __restrict__ Wph, const float* __restrict__ bph,
                          const float* __restrict__ Wpc, const float* __restrict__ bpc,
                          float* __restrict__ h0i, float* __restrict__ c0i) {
  int idx = blockIdx.x * blockDim.x + threadIdx.x;
  if (idx >= L_ * B_ * H_) return;
  int h = idx % H_; int lb = idx / H_;
  const float* eh = encH + lb * U_;
  const float* ec = encC + lb * U_;
  const float* wh = Wph + h * U_;
  const float* wc = Wpc + h * U_;
  float sh = bph[h], sc = bpc[h];
  for (int u = 0; u < U_; ++u) { sh += eh[u] * wh[u]; sc += ec[u] * wc[u]; }
  h0i[idx] = sh; c0i[idx] = sc;
}

// ctx[b,t1,h] = encoder_outs[t1,b,:] @ Wph[h,:] + bph[h]
__global__ void k_ctx(const float* __restrict__ encO, const float* __restrict__ Wph,
                      const float* __restrict__ bph, float* __restrict__ ctxb) {
  int idx = blockIdx.x * blockDim.x + threadIdx.x;
  if (idx >= B_ * T1_ * H_) return;
  int h = idx % H_; int bt = idx / H_;
  int t1 = bt % T1_; int b = bt / T1_;
  const float* eo = encO + (t1 * B_ + b) * U_;
  const float* w  = Wph + h * U_;
  float s = bph[h];
  for (int u = 0; u < U_; ++u) s += eo[u] * w[u];
  ctxb[idx] = s;  // layout (b*T1+t1)*H + h
}

// gctx[b,t1,g] = ctx[b,t1,:] @ Wih0[g, E:E+H]   (k-invariant part of gates)
__global__ void k_gctx(const float* __restrict__ ctxb, const float* __restrict__ Wih0,
                       float* __restrict__ gctx) {
  int idx = blockIdx.x * blockDim.x + threadIdx.x;
  if (idx >= B_ * T1_ * FH_) return;
  int g = idx % FH_; int bt = idx / FH_;
  const float* cx = ctxb + bt * H_;
  const float* w  = Wih0 + g * (H_ + E_) + E_;
  float s = 0.f;
  for (int h = 0; h < H_; ++h) s += cx[h] * w[h];
  gctx[idx] = s;
}

// Per-step embedding part: gembS[b,j,g] = emb(in_tok) @ Wih0[g,:E] + bih0+bhh0
__global__ void k_gemb(const int* __restrict__ prev, const float* __restrict__ embed,
                       const float* __restrict__ Wih0, const float* __restrict__ bih0,
                       const float* __restrict__ bhh0, float* __restrict__ gembS,
                       int kstep) {
  int idx = blockIdx.x * blockDim.x + threadIdx.x;
  if (idx >= B_ * J_ * FH_) return;
  int g = idx % FH_; int bj = idx / FH_;
  int j = bj % J_; int b = bj / J_;
  int tok;
  if (kstep == 0) tok = START_ID_;
  else {
    int p = j + kstep - 1;
    if (p > T2_ - 1) p = T2_ - 1;
    tok = prev[b * T2_ + p];
  }
  const float* em = embed + tok * E_;
  const float* w  = Wih0 + g * (H_ + E_);
  float s = bih0[g] + bhh0[g];
  for (int e = 0; e < E_; ++e) s += em[e] * w[e];
  gembS[idx] = s;
}

__global__ void k_init_state(const float* __restrict__ h0i, const float* __restrict__ c0i,
                             __bf16* __restrict__ h0bf, __bf16* __restrict__ h1bf,
                             float* __restrict__ c0s, float* __restrict__ c1s,
                             float* __restrict__ cum) {
  int idx = blockIdx.x * blockDim.x + threadIdx.x;
  if (idx >= N_ * H_) return;
  int h = idx % H_; int n = idx / H_;
  int b = n / (T1_ * J_);
  h0bf[idx] = (__bf16)h0i[(0 * B_ + b) * H_ + h];
  h1bf[idx] = (__bf16)h0i[(1 * B_ + b) * H_ + h];
  c0s[idx]  = c0i[(0 * B_ + b) * H_ + h];
  c1s[idx]  = c0i[(1 * B_ + b) * H_ + h];
  if (h == 0) cum[n] = 0.f;
}

// ---------------------------------------------------------------------------
extern "C" void kernel_launch(void* const* d_in, const int* in_sizes, int n_in,
                              void* d_out, int out_size, void* d_ws, size_t ws_size,
                              hipStream_t stream) {
  (void)in_sizes; (void)n_in; (void)out_size; (void)ws_size;

  const int*   prev  = (const int*)d_in[0];
  const float* encO  = (const float*)d_in[1];
  const float* encH  = (const float*)d_in[2];
  const float* encC  = (const float*)d_in[3];
  const float* embed = (const float*)d_in[4];
  const float* Wph   = (const float*)d_in[5];
  const float* bph   = (const float*)d_in[6];
  const float* Wpc   = (const float*)d_in[7];
  const float* bpc   = (const float*)d_in[8];
  const float* Wih0  = (const float*)d_in[9];
  const float* Whh0  = (const float*)d_in[10];
  const float* bih0  = (const float*)d_in[11];
  const float* bhh0  = (const float*)d_in[12];
  const float* Wih1  = (const float*)d_in[13];
  const float* Whh1  = (const float*)d_in[14];
  const float* bih1  = (const float*)d_in[15];
  const float* bhh1  = (const float*)d_in[16];
  const float* Wfc   = (const float*)d_in[17];
  const float* bfc   = (const float*)d_in[18];
  const float* Wout  = (const float*)d_in[19];
  const float* bout  = (const float*)d_in[20];
  float* outp = (float*)d_out;

  char* base = (char*)d_ws;
  size_t off = 0;
  auto alloc = [&](size_t bytes) -> char* {
    char* r = base + off;
    off += (bytes + 255) & ~(size_t)255;
    return r;
  };

  float*  c0s    = (float*) alloc((size_t)N_ * H_ * 4);
  float*  c1s    = (float*) alloc((size_t)N_ * H_ * 4);
  __bf16* h0A    = (__bf16*)alloc((size_t)N_ * H_ * 2);
  __bf16* h0B    = (__bf16*)alloc((size_t)N_ * H_ * 2);
  __bf16* h1A    = (__bf16*)alloc((size_t)N_ * H_ * 2);
  __bf16* h1B    = (__bf16*)alloc((size_t)N_ * H_ * 2);
  __bf16* obf    = (__bf16*)alloc((size_t)N_ * OUT_ * 2);
  float*  gembS  = (float*) alloc((size_t)B_ * J_ * FH_ * 4);
  float*  gctx   = (float*) alloc((size_t)B_ * T1_ * FH_ * 4);
  float*  ctxb   = (float*) alloc((size_t)B_ * T1_ * H_ * 4);
  float*  h0i    = (float*) alloc((size_t)L_ * B_ * H_ * 4);
  float*  c0i    = (float*) alloc((size_t)L_ * B_ * H_ * 4);
  float*  cum    = (float*) alloc((size_t)N_ * 4);
  float*  bias1  = (float*) alloc((size_t)FH_ * 4);
  __bf16* Whh0bf = (__bf16*)alloc((size_t)FH_ * H_ * 2);
  __bf16* Wih1bf = (__bf16*)alloc((size_t)FH_ * H_ * 2);
  __bf16* Whh1bf = (__bf16*)alloc((size_t)FH_ * H_ * 2);
  __bf16* Wfcbf  = (__bf16*)alloc((size_t)OUT_ * H_ * 2);
  __bf16* Woutbf = (__bf16*)alloc((size_t)V_ * OUT_ * 2);

  // One-time conversions & projections.
  k_cvt<<<CDIV(FH_ * H_, 256), 256, 0, stream>>>(Whh0, Whh0bf, FH_ * H_);
  k_cvt<<<CDIV(FH_ * H_, 256), 256, 0, stream>>>(Wih1, Wih1bf, FH_ * H_);
  k_cvt<<<CDIV(FH_ * H_, 256), 256, 0, stream>>>(Whh1, Whh1bf, FH_ * H_);
  k_cvt<<<CDIV(OUT_ * H_, 256), 256, 0, stream>>>(Wfc, Wfcbf, OUT_ * H_);
  k_cvt<<<CDIV(V_ * OUT_, 256), 256, 0, stream>>>(Wout, Woutbf, V_ * OUT_);
  k_bias1<<<CDIV(FH_, 256), 256, 0, stream>>>(bih1, bhh1, bias1);
  k_proj_hc<<<CDIV(L_ * B_ * H_, 256), 256, 0, stream>>>(encH, encC, Wph, bph, Wpc, bpc, h0i, c0i);
  k_ctx<<<CDIV(B_ * T1_ * H_, 256), 256, 0, stream>>>(encO, Wph, bph, ctxb);
  k_gctx<<<CDIV(B_ * T1_ * FH_, 256), 256, 0, stream>>>(ctxb, Wih0, gctx);
  k_init_state<<<CDIV(N_ * H_, 256), 256, 0, stream>>>(h0i, c0i, h0A, h1A, c0s, c1s, cum);

  __bf16 *h0cur = h0A, *h0nxt = h0B, *h1cur = h1A, *h1nxt = h1B;
  for (int k = 0; k < KSTEPS_; ++k) {
    k_gemb<<<CDIV(B_ * J_ * FH_, 256), 256, 0, stream>>>(prev, embed, Wih0, bih0, bhh0, gembS, k);
    k_gates<<<dim3(N_ / 64, H_ / 16), 128, 0, stream>>>(
        h0cur, H_, (const __bf16*)nullptr, 0, Whh0bf, (const __bf16*)nullptr,
        gembS, gctx, c0s, h0nxt, 0);
    k_gates<<<dim3(N_ / 64, H_ / 16), 128, 0, stream>>>(
        h0nxt, H_, h1cur, H_, Wih1bf, Whh1bf,
        bias1, (const float*)nullptr, c1s, h1nxt, 1);
    k_fc<<<dim3(N_ / 64, OUT_ / 64), 128, 0, stream>>>(h1nxt, Wfcbf, bfc, obf);
    k_logits<<<dim3(N_ / 64), 128, 0, stream>>>(obf, Woutbf, bout, prev, cum, outp, k);
    __bf16* t;
    t = h0cur; h0cur = h0nxt; h0nxt = t;
    t = h1cur; h1cur = h1nxt; h1nxt = t;
  }
}
